// llstm_35682588295394
// MI455X (gfx1250) — matrix operand order repaired and checked by reading, not compiled, not run on hardware
//
#include <hip/hip_runtime.h>
#include <hip/hip_bf16.h>

// ---------------------------------------------------------------------------
// CDNA5 (gfx1250) bf16 WMMA LSTM pipeline, round 3.
//   k_prep : bf16 weight conversion (concat [W_hh|W_ch]), bias sum
//   k_wx   : wx[t][g][b] = x[b,:,t] @ W_ih^T  (bf16 transposed; 4 t / block)
//   k_scan : 512-step scan, 8 independent batch-split blocks; 256KB weights
//            pinned in LDS; register-double-buffered wx; hw tanh
//   k_lin1 : hs @ lin_w^T + b, relu, +x  -> y2 (bf16 [b][i][t])
//   k_lin2 : y2 @ lin2_w^T + b2, relu   -> out (fp32)
// ---------------------------------------------------------------------------

typedef __attribute__((ext_vector_type(16))) __bf16 v16bf;
typedef __attribute__((ext_vector_type(8)))  __bf16 v8bf;
typedef __attribute__((ext_vector_type(2)))  __bf16 bf2;
typedef __attribute__((ext_vector_type(8)))  float  v8f;

#define BSZ   128
#define II    256      // C*F
#define TT    512
#define HH    256
#define G4    1024     // 4*H
#define KR    512      // recurrent K = 2*H  ([h|c])
#define KL    128      // K slice of wcat pinned in LDS (256 KB)

__device__ __forceinline__ v8f zero_v8f() {
  v8f z;
#pragma unroll
  for (int i = 0; i < 8; ++i) z[i] = 0.0f;
  return z;
}

__device__ __forceinline__ float sigmoidf_(float x) {
  return 1.0f / (1.0f + __expf(-x));
}

__device__ __forceinline__ float fast_tanh(float x) {
#if __has_builtin(__builtin_amdgcn_tanhf)
  return __builtin_amdgcn_tanhf(x);            // v_tanh_f32 (gfx1250 TRANS)
#else
  float ax = fabsf(x);
  float e  = __expf(-2.0f * ax);               // overflow-safe
  float r  = (1.0f - e) / (1.0f + e);
  return copysignf(r, x);
#endif
}

// A fragment (16x32 bf16): lane&15 = row M; half = lane>>4; k = 16*half + s.
__device__ __forceinline__ v16bf a_frag(const __bf16* Arow, int kbase, int half) {
  return *(const v16bf*)(Arow + kbase + 16 * half);
}

// B fragment (32x16 bf16): one aligned 32-byte load from weight row n.
__device__ __forceinline__ v16bf b_frag(const __bf16* Wn, int kbase, int half) {
  return *(const v16bf*)(Wn + kbase + 16 * half);
}

__device__ __forceinline__ v8f wmma_bf16(v16bf a, v16bf b, v8f c) {
  return __builtin_amdgcn_wmma_f32_16x16x32_bf16(
      false, a, false, b, (short)0, c, false, false);
}

// ---------------------------------------------------------------------------
__global__ __launch_bounds__(256) void k_prep(
    const float* __restrict__ w_ih, const float* __restrict__ w_hh,
    const float* __restrict__ w_ch, const float* __restrict__ b_ih,
    const float* __restrict__ b_hh, const float* __restrict__ b_ch,
    const float* __restrict__ lin_w, const float* __restrict__ lin2_w,
    __bf16* __restrict__ wih_bf, __bf16* __restrict__ wcat_bf,
    __bf16* __restrict__ linw_bf, __bf16* __restrict__ lin2_bf,
    float* __restrict__ bias) {
  int idx = blockIdx.x * 256 + threadIdx.x;
  if (idx < G4 * II)  wih_bf[idx]  = (__bf16)w_ih[idx];
  if (idx < G4 * KR) {
    int n = idx >> 9, k = idx & 511;
    float v = (k < HH) ? w_hh[n * HH + k] : w_ch[n * HH + (k - HH)];
    wcat_bf[idx] = (__bf16)v;
  }
  if (idx < II * HH)  linw_bf[idx] = (__bf16)lin_w[idx];
  if (idx < TT * TT)  lin2_bf[idx] = (__bf16)lin2_w[idx];
  if (idx < G4)       bias[idx]    = b_ih[idx] + b_hh[idx] + b_ch[idx];
}

// ---------------------------------------------------------------------------
// wx GEMM: wx[t][g][b]. 4 consecutive t per block: float4 x reads along t,
// 4x B-fragment reuse. grid (T/4, BS/16, 1024/128).
// ---------------------------------------------------------------------------
__global__ __launch_bounds__(256) void k_wx(
    const float* __restrict__ x, const __bf16* __restrict__ wih,
    __bf16* __restrict__ wx) {
  __shared__ __bf16 As[4][16][II];  // 32 KB
  const int t0 = blockIdx.x * 4;
  const int b0 = blockIdx.y * 16;
  const int wave = threadIdx.x >> 5, lane = threadIdx.x & 31;
  const int half = lane >> 4, ln = lane & 15;
  const int n0 = blockIdx.z * 128 + wave * 16;

  for (int idx = threadIdx.x; idx < 16 * II; idx += 256) {
    int m = idx >> 8, i = idx & 255;
    float4 xv = *(const float4*)(x + ((size_t)(b0 + m) * II + i) * TT + t0);
    As[0][m][i] = (__bf16)xv.x;
    As[1][m][i] = (__bf16)xv.y;
    As[2][m][i] = (__bf16)xv.z;
    As[3][m][i] = (__bf16)xv.w;
  }
  __syncthreads();

  v8f acc[4];
#pragma unroll
  for (int ts = 0; ts < 4; ++ts) acc[ts] = zero_v8f();
#pragma unroll
  for (int k = 0; k < II; k += 32) {
    v16bf b = b_frag(wih + (size_t)(n0 + ln) * II, k, half);
#pragma unroll
    for (int ts = 0; ts < 4; ++ts) {
      v16bf a = a_frag(&As[ts][ln][0], k, half);
      acc[ts] = wmma_bf16(a, b, acc[ts]);
    }
  }
#pragma unroll
  for (int ts = 0; ts < 4; ++ts) {
    v8bf o;
#pragma unroll
    for (int r = 0; r < 8; ++r) o[r] = (__bf16)acc[ts][r];
    *(v8bf*)(wx + ((size_t)(t0 + ts) * G4 + n0 + ln) * BSZ + b0 + 8 * half) = o;
  }
}

// ---------------------------------------------------------------------------
// LSTM scan
// ---------------------------------------------------------------------------
struct WxRegs { v8bf v[4][2]; };

__global__ __launch_bounds__(256) void k_scan(
    const __bf16* __restrict__ wx,    // [T][1024][BS]
    const __bf16* __restrict__ wcat,  // [1024][512]  ([W_hh|W_ch])
    const float* __restrict__ bias,   // [1024] summed
    const float* __restrict__ h0, const float* __restrict__ c0,
    __bf16* __restrict__ hs) {        // [T][BS][H]
  __shared__ __bf16 wlds[G4][KL];  // 256 KB : wcat[:, 0:128]
  __shared__ __bf16 hc[16][KR];    //  16 KB : cols 0..255 h, 256..511 c
  __shared__ float  cst[16][HH];   //  16 KB : fp32 cell state

  const int b0 = blockIdx.x * 16;
  const int tid = threadIdx.x;
  const int wave = tid >> 5, lane = tid & 31;
  const int half = lane >> 4, ln = lane & 15;
  const int jbase = wave * 32;

  for (int i = tid; i < G4 * KL / 8; i += 256) {
    int n = i >> 4, q = i & 15;
    ((uint4*)&wlds[n][0])[q] = ((const uint4*)(wcat + (size_t)n * KR))[q];
  }
  for (int idx = tid; idx < 16 * HH; idx += 256) {
    int m = idx >> 8, j = idx & 255;
    float hv = h0[(b0 + m) * HH + j];
    float cv = c0[(b0 + m) * HH + j];
    hc[m][j]      = (__bf16)hv;
    hc[m][HH + j] = (__bf16)cv;
    cst[m][j]     = cv;
  }
  __syncthreads();

  auto load_wx = [&](int t, WxRegs& w) {
#pragma unroll
    for (int g = 0; g < 4; ++g)
#pragma unroll
      for (int s = 0; s < 2; ++s) {
        int n = g * HH + jbase + s * 16 + ln;
        w.v[g][s] =
            *(const v8bf*)(wx + ((size_t)t * G4 + n) * BSZ + b0 + 8 * half);
      }
  };

  auto step = [&](int t, const WxRegs& wcur, WxRegs& wnext) {
    // --- acc init from pre-loaded registers: z = wx_t + bias ---
    v8f acc[4][2];
#pragma unroll
    for (int g = 0; g < 4; ++g)
#pragma unroll
      for (int s = 0; s < 2; ++s) {
        int n = g * HH + jbase + s * 16 + ln;
        float bn = bias[n];
        v8f c;
#pragma unroll
        for (int r = 0; r < 8; ++r) c[r] = bn + (float)wcur.v[g][s][r];
        acc[g][s] = c;
      }

    // --- issue next step's wx loads early (hidden under the GEMM) ---
    int tn = (t + 1 < TT) ? t + 1 : TT - 1;
    load_wx(tn, wnext);
    // --- L2 streaming prefetch two steps ahead (1 line / thread / 4 rows) ---
    int tp = (t + 2 < TT) ? t + 2 : TT - 1;
#pragma unroll
    for (int q = 0; q < 4; ++q)
      __builtin_prefetch(wx + ((size_t)tp * G4 + tid + q * 256) * BSZ + b0, 0, 1);

    // --- z += [h|c] @ [W_hh|W_ch]^T : K<128 from LDS, rest from L2 ---
#pragma unroll 2
    for (int k = 0; k < KL; k += 32) {
      v16bf a = a_frag(&hc[ln][0], k, half);
#pragma unroll
      for (int g = 0; g < 4; ++g)
#pragma unroll
        for (int s = 0; s < 2; ++s) {
          int n = g * HH + jbase + s * 16 + ln;
          acc[g][s] = wmma_bf16(a, b_frag(&wlds[n][0], k, half), acc[g][s]);
        }
    }
#pragma unroll 2
    for (int k = KL; k < KR; k += 32) {
      v16bf a = a_frag(&hc[ln][0], k, half);
#pragma unroll
      for (int g = 0; g < 4; ++g)
#pragma unroll
        for (int s = 0; s < 2; ++s) {
          int n = g * HH + jbase + s * 16 + ln;
          acc[g][s] = wmma_bf16(a, b_frag(wcat + (size_t)n * KR, k, half), acc[g][s]);
        }
    }
    __syncthreads();  // all reads of hc for step t done

    // --- gates + cell update in-register ---
#pragma unroll
    for (int s = 0; s < 2; ++s) {
      int j = jbase + s * 16 + ln;
#pragma unroll
      for (int r = 0; r < 8; ++r) {
        int m = r + 8 * half;
        float iv = sigmoidf_(acc[0][s][r]);
        float fv = sigmoidf_(acc[1][s][r]);
        float gv = fast_tanh(acc[2][s][r]);
        float ov = sigmoidf_(acc[3][s][r]);
        float cp = cst[m][j];
        float cn = fv * cp + iv * gv;
        float hn = ov * fast_tanh(cn);
        cst[m][j]     = cn;
        hc[m][HH + j] = (__bf16)cn;
        hc[m][j]      = (__bf16)hn;
      }
    }
    __syncthreads();  // state visible

    // --- coalesced hs export (ordered before next gate-write by the next
    //     iteration's post-GEMM barrier) ---
    for (int i = tid; i < 512; i += 256) {
      int m = i >> 5, q = i & 31;
      ((uint4*)(hs + ((size_t)t * BSZ + b0 + m) * HH))[q] =
          ((const uint4*)&hc[m][0])[q];
    }
  };

  WxRegs wa, wb;
  load_wx(0, wa);
  for (int t = 0; t < TT; t += 2) {
    step(t, wa, wb);
    step(t + 1, wb, wa);
  }
}

// ---------------------------------------------------------------------------
// lin1: y2[b][i][t] = relu(hs[t,b,:]@lin_w[i,:]+lin_b[i]) + x[b,i,t]
// grid (BS, T/16): D rows run along t -> vector x reads, packed bf16 stores.
// ---------------------------------------------------------------------------
__global__ __launch_bounds__(256) void k_lin1(
    const __bf16* __restrict__ hs, const __bf16* __restrict__ linw,
    const float* __restrict__ linb, const float* __restrict__ x,
    __bf16* __restrict__ y2) {
  __shared__ __bf16 As[16][HH];  // 8 KB
  const int b  = blockIdx.x;
  const int t0 = blockIdx.y * 16;
  for (int i = threadIdx.x; i < 512; i += 256) {
    int m = i >> 5, q = i & 31;
    ((uint4*)&As[m][0])[q] =
        ((const uint4*)(hs + ((size_t)(t0 + m) * BSZ + b) * HH))[q];
  }
  __syncthreads();
  const int wave = threadIdx.x >> 5, lane = threadIdx.x & 31;
  const int half = lane >> 4, ln = lane & 15;

  v8f acc0 = zero_v8f(), acc1 = zero_v8f();
  const int na = wave * 32 + ln, nb = wave * 32 + 16 + ln;
#pragma unroll
  for (int k = 0; k < HH; k += 32) {
    v16bf a = a_frag(&As[ln][0], k, half);
    acc0 = wmma_bf16(a, b_frag(linw + (size_t)na * HH, k, half), acc0);
    acc1 = wmma_bf16(a, b_frag(linw + (size_t)nb * HH, k, half), acc1);
  }
#pragma unroll
  for (int s = 0; s < 2; ++s) {
    int n = (s == 0) ? na : nb;
    v8f acc = (s == 0) ? acc0 : acc1;
    float bn = linb[n];
    size_t off = ((size_t)b * II + n) * TT + t0 + 8 * half;
    v8f xv = *(const v8f*)(x + off);
    v8bf o;
#pragma unroll
    for (int r = 0; r < 8; ++r) {
      float v = acc[r] + bn;
      v = v > 0.0f ? v : 0.0f;
      o[r] = (__bf16)(v + xv[r]);
    }
    *(v8bf*)(y2 + off) = o;
  }
}

// ---------------------------------------------------------------------------
// lin2: out = relu(y2 @ lin2_w^T + b2); output staged via LDS -> b128 stores.
// ---------------------------------------------------------------------------
__global__ __launch_bounds__(256) void k_lin2(
    const __bf16* __restrict__ y2, const __bf16* __restrict__ w2,
    const float* __restrict__ b2, float* __restrict__ out) {
  __shared__ __bf16 As[16][TT];  // 16 KB
  __shared__ float  Os[16][TT];  // 32 KB
  const int m0 = blockIdx.x * 16;
  {
    const uint4* src = (const uint4*)(y2 + (size_t)m0 * TT);
    uint4* dst = (uint4*)(&As[0][0]);
    for (int i = threadIdx.x; i < 16 * TT / 8; i += 256) dst[i] = src[i];
  }
  __syncthreads();
  const int wave = threadIdx.x >> 5, lane = threadIdx.x & 31;
  const int half = lane >> 4, ln = lane & 15;

  v8f acc[4];
#pragma unroll
  for (int s = 0; s < 4; ++s) acc[s] = zero_v8f();

#pragma unroll 4
  for (int k = 0; k < TT; k += 32) {
    v16bf a = a_frag(&As[ln][0], k, half);
#pragma unroll
    for (int s = 0; s < 4; ++s) {
      int n = wave * 64 + s * 16 + ln;
      acc[s] = wmma_bf16(a, b_frag(w2 + (size_t)n * TT, k, half), acc[s]);
    }
  }
#pragma unroll
  for (int s = 0; s < 4; ++s) {
    int n = wave * 64 + s * 16 + ln;
    float bn = b2[n];
#pragma unroll
    for (int r = 0; r < 8; ++r) {
      float v = acc[s][r] + bn;
      Os[r + 8 * half][n] = v > 0.0f ? v : 0.0f;
    }
  }
  __syncthreads();
  for (int i = threadIdx.x; i < 16 * TT / 4; i += 256) {
    int m = i >> 7, q = i & 127;
    ((uint4*)(out + (size_t)(m0 + m) * TT))[q] = ((const uint4*)&Os[m][0])[q];
  }
}

// ---------------------------------------------------------------------------
extern "C" void kernel_launch(void* const* d_in, const int* in_sizes, int n_in,
                              void* d_out, int out_size, void* d_ws, size_t ws_size,
                              hipStream_t stream) {
  const float* x     = (const float*)d_in[0];
  const float* w_ih  = (const float*)d_in[1];
  const float* w_hh  = (const float*)d_in[2];
  const float* w_ch  = (const float*)d_in[3];
  const float* b_ih  = (const float*)d_in[4];
  const float* b_hh  = (const float*)d_in[5];
  const float* b_ch  = (const float*)d_in[6];
  const float* h0    = (const float*)d_in[7];
  const float* c0    = (const float*)d_in[8];
  const float* lin_w = (const float*)d_in[9];
  const float* lin_b = (const float*)d_in[10];
  const float* lin2_w= (const float*)d_in[11];
  const float* lin2_b= (const float*)d_in[12];
  float* out = (float*)d_out;

  char* p = (char*)d_ws;
  __bf16* wx_buf   = (__bf16*)p; p += (size_t)TT * G4 * BSZ * 2;   // 128 MB
  __bf16* hs_buf   = (__bf16*)p; p += (size_t)TT * BSZ * HH * 2;   //  32 MB
  __bf16* y2_buf   = (__bf16*)p; p += (size_t)BSZ * II * TT * 2;   //  32 MB
  __bf16* wih_bf   = (__bf16*)p; p += (size_t)G4 * II * 2;
  __bf16* wcat_bf  = (__bf16*)p; p += (size_t)G4 * KR * 2;
  __bf16* linw_bf  = (__bf16*)p; p += (size_t)II * HH * 2;
  __bf16* lin2_bf  = (__bf16*)p; p += (size_t)TT * TT * 2;
  float*  bias_buf = (float*)p;  p += (size_t)G4 * 4;
  (void)ws_size; (void)in_sizes; (void)n_in; (void)out_size;

  k_prep<<<(G4 * KR + 255) / 256, 256, 0, stream>>>(
      w_ih, w_hh, w_ch, b_ih, b_hh, b_ch, lin_w, lin2_w,
      wih_bf, wcat_bf, linw_bf, lin2_bf, bias_buf);

  k_wx<<<dim3(TT / 4, BSZ / 16, G4 / 128), 256, 0, stream>>>(x, wih_bf, wx_buf);

  k_scan<<<BSZ / 16, 256, 0, stream>>>(wx_buf, wcat_bf, bias_buf, h0, c0, hs_buf);

  k_lin1<<<dim3(BSZ, TT / 16), 256, 0, stream>>>(hs_buf, linw_bf, lin_b, x, y2_buf);

  k_lin2<<<(BSZ * II) / 16, 256, 0, stream>>>(y2_buf, lin2_bf, lin2_b, out);
}
